// DeepST_model_77103252897901
// MI455X (gfx1250) — compile-verified, matrix operand order checked by reading, not run on hardware
//
#include <hip/hip_runtime.h>
#include <math.h>

// ---------------------------------------------------------------------------
// DeepST forward on gfx1250 (MI455X).  Memory-bound: x read (524MB) + de_feat
// write (524MB) dominate -> ~45us HBM floor.  Big GEMMs use v_wmma_f32_16x16x32_f16
// (fp32 accum); encoder B staged via GLOBAL_LOAD_ASYNC_TO_LDS, decoder B staged
// via double-buffered TDM (tensor_load_to_lds + s_wait_tensorcnt 1 pipeline);
// GCN scatter uses fp32 atomics on L2-resident intermediates.
// ---------------------------------------------------------------------------

typedef __attribute__((ext_vector_type(16))) _Float16 v16h;
typedef __attribute__((ext_vector_type(8)))  float    v8f;
typedef __attribute__((ext_vector_type(4)))  unsigned u32x4;
typedef __attribute__((ext_vector_type(8)))  int      i32x8;
typedef __attribute__((ext_vector_type(4)))  int      i32x4;

enum : int {
  N     = 65536,
  DIN   = 2000,
  E     = 1048576,
  ENC0  = 32,
  ENC1  = 20,
  CONV0 = 64,
  ZG    = 8,
  ZDIM  = 28,     // ENC1 + ZG
  DEC0  = 32,
  NC    = 15,
  KB_ENC = 63,    // ceil(2000/32); kb 0..61 full, kb 62 = 16-wide tail
  CT_DEC = 125,   // 2000/16
  KCH    = 8,     // encoder K-chunk staged in LDS (8*1024 halves = 16KB)
  CCH    = 25,    // decoder col-tile chunk (25*512 halves = 25.6KB per buffer)
  NCHUNK = 5,     // CT_DEC / CCH
};

// ---- gfx1250 async / TDM feature gating (device pass only) ----------------
#if defined(__HIP_DEVICE_COMPILE__)
  #if __has_builtin(__builtin_amdgcn_global_load_async_to_lds_b128)
    #define HAS_ASYNC 1
  #else
    #define HAS_ASYNC 0
  #endif
  #if __has_builtin(__builtin_amdgcn_tensor_load_to_lds)
    #define HAS_TDM 1
  #else
    #define HAS_TDM 0
  #endif
  #if __has_builtin(__builtin_amdgcn_s_wait_asynccnt)
    #define WAIT_ASYNC() __builtin_amdgcn_s_wait_asynccnt(0)
  #elif HAS_ASYNC
    #define WAIT_ASYNC() asm volatile("s_wait_asynccnt 0x0" ::: "memory")
  #else
    #define WAIT_ASYNC() ((void)0)
  #endif
#else
  #define HAS_ASYNC 0
  #define HAS_TDM 0
  #define WAIT_ASYNC() ((void)0)
#endif

__device__ __forceinline__ float elu1(float v) { return v > 0.0f ? v : (expf(v) - 1.0f); }

// per-lane 16B global -> LDS copy (async engine when available)
__device__ __forceinline__ void stage16(const _Float16* g, _Float16* l) {
#if HAS_ASYNC
  __builtin_amdgcn_global_load_async_to_lds_b128(
      (__attribute__((address_space(1))) i32x4*)(unsigned long long)(const void*)g,
      (__attribute__((address_space(3))) i32x4*)(__attribute__((address_space(3))) void*)l,
      0, 0);
#else
  *(uint4*)l = *(const uint4*)g;
#endif
}

// byte offset of an LDS object within the wave's LDS allocation
__device__ __forceinline__ unsigned lds_off(void* p) {
#if defined(__HIP_DEVICE_COMPILE__)
  return (unsigned)(unsigned long long)(__attribute__((address_space(3))) char*)p;
#else
  (void)p; return 0;
#endif
}

// Stage one decoder B chunk (CCH*512 halves = 25.6KB contiguous) into LDS.
// TDM path: wave 0 builds a 1D D# (data_size=2B, tensor_dim0=tile_dim0=12800,
// stride=12800, type=2) and issues tensor_load_to_lds (tracked by TENSORcnt).
__device__ __forceinline__ void stage_dec_chunk(const _Float16* srcB, _Float16* ldst,
                                                int tid) {
#if HAS_TDM
  if (tid < 32) {
    const unsigned long long ga = (unsigned long long)(const void*)srcB;
    const unsigned ldsa = lds_off(ldst);
    enum : unsigned { NH = CCH * 512u };          // halves in tile = 12800
    u32x4 g0 = { 1u,                               // count=1 (valid D#)
                 ldsa,                             // lds_addr
                 (unsigned)(ga & 0xffffffffu),     // global_addr[31:0]
                 (unsigned)((ga >> 32) & 0x1ffffffu) | (2u << 30) }; // [56:32]+type=2
    i32x8 g1 = { (int)(1u << 16),                  // data_size=1 (2 bytes)
                 (int)((NH & 0xffffu) << 16),      // tensor_dim0[15:0]
                 (int)((NH >> 16) | (1u << 16)),   // tensor_dim0[31:16], tensor_dim1=1
                 (int)((NH & 0xffffu) << 16),      // tile_dim0
                 0,                                // tile_dim1=0, tile_dim2=0 (1D)
                 (int)NH,                          // tensor_dim0_stride[31:0]
                 0, 0 };
    i32x4 gz = { 0, 0, 0, 0 };
#if __clang_major__ >= 23
    i32x8 gz8 = {};
    __builtin_amdgcn_tensor_load_to_lds(g0, g1, gz, gz, gz8, 0);
#else
    __builtin_amdgcn_tensor_load_to_lds(g0, g1, gz, gz, 0);
#endif
  }
#else
  for (int off = tid * 8; off < CCH * 512; off += 256 * 8)
    stage16(srcB + off, ldst + off);
#endif
}

// ---------------------------------------------------------------------------
// Degree / normalization
// ---------------------------------------------------------------------------
__global__ void k_deg_init(float* __restrict__ deg) {
  int i = blockIdx.x * blockDim.x + threadIdx.x;
  if (i < N) deg[i] = 1.0f;               // self-loop contribution
}

__global__ void k_deg_acc(const int* __restrict__ ei, float* __restrict__ deg) {
  int e = blockIdx.x * blockDim.x + threadIdx.x;
  if (e < E) atomicAdd(&deg[ei[E + e]], 1.0f);   // dst = ei[1][e]
}

__global__ void k_dinv(float* __restrict__ deg) {
  int i = blockIdx.x * blockDim.x + threadIdx.x;
  if (i < N) { float d = deg[i]; deg[i] = d > 0.0f ? rsqrtf(d) : 0.0f; }
}

// ---------------------------------------------------------------------------
// Weight packing into WMMA B-fragment order (16-bit B 32x16, wave32 layout):
// lane<16 : col n = lane,    k_local = {0..7, 16..23}
// lane>=16: col n = lane-16, k_local = {8..15, 24..31}
// flattened: Wpk[((kb*2+ct)*32 + lane)*16 + i]
// ---------------------------------------------------------------------------
__global__ void k_prep_enc_w(const float* __restrict__ eW0, _Float16* __restrict__ Wb) {
  int e = blockIdx.x * blockDim.x + threadIdx.x;
  if (e >= KB_ENC * 2 * 32 * 16) return;
  int i    = e & 15;
  int lane = (e >> 4) & 31;
  int ctkb = e >> 9;
  int ct   = ctkb & 1;
  int kb   = ctkb >> 1;
  int half = lane >> 4;
  int n    = ct * 16 + (lane & 15);
  int kl   = (i < 8) ? (i + 8 * half) : (i + 8 + 8 * half);
  int k    = kb * 32 + kl;
  Wb[e] = (_Float16)((k < DIN) ? eW0[k * ENC0 + n] : 0.0f);
}

__global__ void k_prep_dec_w(const float* __restrict__ dW1, _Float16* __restrict__ Wd) {
  int e = blockIdx.x * blockDim.x + threadIdx.x;
  if (e >= CT_DEC * 32 * 16) return;
  int i    = e & 15;
  int lane = (e >> 4) & 31;
  int ct   = e >> 9;                       // 0..124
  int half = lane >> 4;
  int n    = ct * 16 + (lane & 15);
  int kl   = (i < 8) ? (i + 8 * half) : (i + 8 + 8 * half);
  Wd[e] = (_Float16)dW1[kl * DIN + n];
}

// ---------------------------------------------------------------------------
// Encoder L0: H = elu(bn(X @ eW0 + eb0)).  WMMA f16, fp32 accum.
// kb 0..61 are branch-free (k<=1999 statically in range): four float4 loads
// per lane + packed cvt; kb 62 peeled with compile-time zero upper half.
// B fragments staged per-block into LDS with async-to-LDS in chunks of 8.
// ---------------------------------------------------------------------------
__global__ __launch_bounds__(256) void k_gemm_enc(
    const float* __restrict__ X, const _Float16* __restrict__ Wb,
    const float* __restrict__ eb0, const float* __restrict__ eg0,
    const float* __restrict__ ebe0, float* __restrict__ H) {
  __shared__ _Float16 lBe[KCH * 1024];
  const int lane = threadIdx.x & 31;
  const int wave = threadIdx.x >> 5;
  const int half = lane >> 4;
  const int l16  = lane & 15;
  const size_t mbase = (size_t)blockIdx.x * 128 + wave * 16;
  const float* xr = X + (mbase + l16) * DIN;   // A-frag row for this lane
  v8f acc0 = {}; v8f acc1 = {};
  for (int kc = 0; kc < 62; kc += KCH) {
    const int nk = (62 - kc < KCH) ? (62 - kc) : KCH;
    const _Float16* srcB = Wb + (size_t)kc * 1024;
    for (int off = threadIdx.x * 8; off < nk * 1024; off += 256 * 8)
      stage16(srcB + off, lBe + off);
    WAIT_ASYNC();
    __syncthreads();
    for (int kk = 0; kk < nk; ++kk) {
      const int kbase = (kc + kk) * 32;
      __builtin_prefetch(xr + kbase + 64, 0, 1);
      const float4 f0 = *(const float4*)(xr + kbase + 8 * half);
      const float4 f1 = *(const float4*)(xr + kbase + 8 * half + 4);
      const float4 f2 = *(const float4*)(xr + kbase + 16 + 8 * half);
      const float4 f3 = *(const float4*)(xr + kbase + 20 + 8 * half);
      v16h a;
      a[0]  = (_Float16)f0.x; a[1]  = (_Float16)f0.y;
      a[2]  = (_Float16)f0.z; a[3]  = (_Float16)f0.w;
      a[4]  = (_Float16)f1.x; a[5]  = (_Float16)f1.y;
      a[6]  = (_Float16)f1.z; a[7]  = (_Float16)f1.w;
      a[8]  = (_Float16)f2.x; a[9]  = (_Float16)f2.y;
      a[10] = (_Float16)f2.z; a[11] = (_Float16)f2.w;
      a[12] = (_Float16)f3.x; a[13] = (_Float16)f3.y;
      a[14] = (_Float16)f3.z; a[15] = (_Float16)f3.w;
      const v16h b0 = *(const v16h*)(lBe + ((size_t)(kk * 2 + 0) * 32 + lane) * 16);
      const v16h b1 = *(const v16h*)(lBe + ((size_t)(kk * 2 + 1) * 32 + lane) * 16);
      acc0 = __builtin_amdgcn_wmma_f32_16x16x32_f16(false, a, false, b0, (short)0, acc0, false, false);
      acc1 = __builtin_amdgcn_wmma_f32_16x16x32_f16(false, a, false, b1, (short)0, acc1, false, false);
    }
    __syncthreads();
  }
  {  // K tail kb=62: k = 1984 + k_local, valid for k_local 0..15 only
    const float4 f0 = *(const float4*)(xr + 1984 + 8 * half);
    const float4 f1 = *(const float4*)(xr + 1984 + 8 * half + 4);
    v16h a;
    a[0] = (_Float16)f0.x; a[1] = (_Float16)f0.y;
    a[2] = (_Float16)f0.z; a[3] = (_Float16)f0.w;
    a[4] = (_Float16)f1.x; a[5] = (_Float16)f1.y;
    a[6] = (_Float16)f1.z; a[7] = (_Float16)f1.w;
#pragma unroll
    for (int i = 8; i < 16; ++i) a[i] = (_Float16)0.0f;
    const v16h b0 = *(const v16h*)(Wb + ((size_t)(62 * 2 + 0) * 32 + lane) * 16);
    const v16h b1 = *(const v16h*)(Wb + ((size_t)(62 * 2 + 1) * 32 + lane) * 16);
    acc0 = __builtin_amdgcn_wmma_f32_16x16x32_f16(false, a, false, b0, (short)0, acc0, false, false);
    acc1 = __builtin_amdgcn_wmma_f32_16x16x32_f16(false, a, false, b1, (short)0, acc1, false, false);
  }
  const float inv = rsqrtf(1.0f + 1e-3f);
#pragma unroll
  for (int r = 0; r < 8; ++r) {
    size_t m = mbase + r + 8 * half;        // C layout: vgpr r -> M=r (+8 for hi half)
    {
      int c = l16;
      float v = acc0[r] + eb0[c];
      v = v * (eg0[c] * inv) + ebe0[c];
      H[m * ENC0 + c] = elu1(v);
    }
    {
      int c = 16 + l16;
      float v = acc1[r] + eb0[c];
      v = v * (eg0[c] * inv) + ebe0[c];
      H[m * ENC0 + c] = elu1(v);
    }
  }
}

// ---------------------------------------------------------------------------
// Encoder L1: feat_x = elu(bn(H @ eW1 + eb1));  writes out_fx and z[:, :20].
// ---------------------------------------------------------------------------
__global__ void k_enc2(const float* __restrict__ H, const float* __restrict__ eW1,
                       const float* __restrict__ eb1, const float* __restrict__ eg1,
                       const float* __restrict__ ebe1,
                       float* __restrict__ out_fx, float* __restrict__ out_z) {
  int row = blockIdx.x * blockDim.x + threadIdx.x;
  if (row >= N) return;
  float h[ENC0];
#pragma unroll
  for (int k = 0; k < ENC0; ++k) h[k] = H[(size_t)row * ENC0 + k];
  const float inv = rsqrtf(1.0f + 1e-3f);
  for (int c = 0; c < ENC1; ++c) {
    float acc = eb1[c];
#pragma unroll
    for (int k = 0; k < ENC0; ++k) acc += h[k] * eW1[k * ENC1 + c];
    acc = elu1(acc * (eg1[c] * inv) + ebe1[c]);
    out_fx[(size_t)row * ENC1 + c] = acc;
    out_z[(size_t)row * ZDIM + c]  = acc;
  }
}

// ---------------------------------------------------------------------------
// GCN layer 1: xw1 = feat_x @ cW  (K=20, 64 out); then self-loop init, edge
// scatter with fp32 atomics (xw1/cxacc are L2-resident), then BN+ReLU.
// ---------------------------------------------------------------------------
__global__ void k_xw1(const float* __restrict__ fx, const float* __restrict__ cW,
                      float* __restrict__ xw1) {
  int t = blockIdx.x * blockDim.x + threadIdx.x;   // N*4 threads: 16 cols each
  if (t >= N * 4) return;
  int row = t >> 2, cg = t & 3;
  float acc[16];
#pragma unroll
  for (int c = 0; c < 16; ++c) acc[c] = 0.0f;
  for (int k = 0; k < ENC1; ++k) {
    float f = fx[(size_t)row * ENC1 + k];
#pragma unroll
    for (int c = 0; c < 16; ++c) acc[c] += f * cW[k * CONV0 + cg * 16 + c];
  }
#pragma unroll
  for (int c = 0; c < 16; ++c) xw1[(size_t)row * CONV0 + cg * 16 + c] = acc[c];
}

__global__ void k_gcn_init(const float* __restrict__ xw1, const float* __restrict__ dinv,
                           float* __restrict__ cxacc) {
  size_t t = (size_t)blockIdx.x * blockDim.x + threadIdx.x;
  if (t >= (size_t)N * CONV0) return;
  int i = (int)(t >> 6);
  float w = dinv[i]; w *= w;
  cxacc[t] = xw1[t] * w;
}

__global__ void k_gcn_edge(const int* __restrict__ ei, const float* __restrict__ dinv,
                           const float* __restrict__ xw1, float* __restrict__ cxacc) {
  int e = blockIdx.x * blockDim.x + threadIdx.x;
  if (e >= E) return;
  int s = ei[e], d = ei[E + e];
  float w = dinv[s] * dinv[d];
  const float4* src = (const float4*)(xw1 + (size_t)s * CONV0);
  float* dst = cxacc + (size_t)d * CONV0;
#pragma unroll
  for (int c4 = 0; c4 < CONV0 / 4; ++c4) {
    float4 v = src[c4];
    atomicAdd(dst + c4 * 4 + 0, v.x * w);
    atomicAdd(dst + c4 * 4 + 1, v.y * w);
    atomicAdd(dst + c4 * 4 + 2, v.z * w);
    atomicAdd(dst + c4 * 4 + 3, v.w * w);
  }
}

__global__ void k_gcn_fin(float* __restrict__ cxacc, const float* __restrict__ cb,
                          const float* __restrict__ cg, const float* __restrict__ cbe) {
  size_t t = (size_t)blockIdx.x * blockDim.x + threadIdx.x;
  if (t >= (size_t)N * CONV0) return;
  int c = (int)(t & 63);
  float v = cxacc[t] + cb[c];
  v = v * (cg[c] * rsqrtf(1.0f + 1e-5f)) + cbe[c];
  cxacc[t] = fmaxf(v, 0.0f);
}

// ---------------------------------------------------------------------------
// GCN heads: xw_m = cx @ mW, xw_l = cx @ lW (K=64, 8 out each); accumulate
// directly into out_mu / out_lv slots; finalize adds bias + fans out to z/gnn_z.
// ---------------------------------------------------------------------------
__global__ void k_heads(const float* __restrict__ cx, const float* __restrict__ mW,
                        const float* __restrict__ lW, float* __restrict__ xwm,
                        float* __restrict__ xwl) {
  int row = blockIdx.x * blockDim.x + threadIdx.x;
  if (row >= N) return;
  const float* xr = cx + (size_t)row * CONV0;
  for (int c = 0; c < ZG; ++c) {
    float am = 0.0f, al = 0.0f;
    for (int k = 0; k < CONV0; ++k) {
      float x = xr[k];
      am += x * mW[k * ZG + c];
      al += x * lW[k * ZG + c];
    }
    xwm[(size_t)row * ZG + c] = am;
    xwl[(size_t)row * ZG + c] = al;
  }
}

__global__ void k_head_init(const float* __restrict__ xwm, const float* __restrict__ xwl,
                            const float* __restrict__ dinv,
                            float* __restrict__ out_mu, float* __restrict__ out_lv) {
  size_t t = (size_t)blockIdx.x * blockDim.x + threadIdx.x;
  if (t >= (size_t)N * ZG) return;
  int i = (int)(t >> 3);
  float w = dinv[i]; w *= w;
  out_mu[t] = xwm[t] * w;
  out_lv[t] = xwl[t] * w;
}

__global__ void k_head_edge(const int* __restrict__ ei, const float* __restrict__ dinv,
                            const float* __restrict__ xwm, const float* __restrict__ xwl,
                            float* __restrict__ out_mu, float* __restrict__ out_lv) {
  int e = blockIdx.x * blockDim.x + threadIdx.x;
  if (e >= E) return;
  int s = ei[e], d = ei[E + e];
  float w = dinv[s] * dinv[d];
#pragma unroll
  for (int c = 0; c < ZG; ++c) {
    atomicAdd(&out_mu[(size_t)d * ZG + c], xwm[(size_t)s * ZG + c] * w);
    atomicAdd(&out_lv[(size_t)d * ZG + c], xwl[(size_t)s * ZG + c] * w);
  }
}

__global__ void k_head_fin(float* __restrict__ out_mu, float* __restrict__ out_lv,
                           const float* __restrict__ mb, const float* __restrict__ lb,
                           float* __restrict__ out_gz, float* __restrict__ out_z) {
  int row = blockIdx.x * blockDim.x + threadIdx.x;
  if (row >= N) return;
#pragma unroll
  for (int c = 0; c < ZG; ++c) {
    float m = out_mu[(size_t)row * ZG + c] + mb[c];
    out_mu[(size_t)row * ZG + c] = m;
    out_gz[(size_t)row * ZG + c] = m;
    out_z[(size_t)row * ZDIM + ENC1 + c] = m;
    out_lv[(size_t)row * ZG + c] += lb[c];
  }
}

// ---------------------------------------------------------------------------
// Decoder L0: d = elu(bn(z @ dW0 + db0))
// ---------------------------------------------------------------------------
__global__ void k_dec0(const float* __restrict__ z, const float* __restrict__ dW0,
                       const float* __restrict__ db0, const float* __restrict__ dg0,
                       const float* __restrict__ dbe0, float* __restrict__ dbuf) {
  size_t t = (size_t)blockIdx.x * blockDim.x + threadIdx.x;
  if (t >= (size_t)N * DEC0) return;
  int row = (int)(t >> 5), c = (int)(t & 31);
  float acc = db0[c];
  const float* zr = z + (size_t)row * ZDIM;
#pragma unroll
  for (int k = 0; k < ZDIM; ++k) acc += zr[k] * dW0[k * DEC0 + c];
  acc = acc * (dg0[c] * rsqrtf(1.0f + 1e-3f)) + dbe0[c];
  dbuf[t] = elu1(acc);
}

// ---------------------------------------------------------------------------
// Decoder final: de_feat = d @ dW1 + db1.  WMMA f16, one K-step (K=32),
// 125 column tiles in 5 chunks of 25.  Double-buffered TDM pipeline: wave 0
// issues tensor_load_to_lds for chunk c+1 before computing chunk c, then
// s_wait_tensorcnt 1 (in-order TENSORcnt => chunk c landed) + barrier.
// ---------------------------------------------------------------------------
__global__ __launch_bounds__(256) void k_gemm_dec(
    const float* __restrict__ D, const _Float16* __restrict__ Wd,
    const float* __restrict__ db1, float* __restrict__ out_de) {
  __shared__ _Float16 lB[2][CCH * 512];
  const int lane = threadIdx.x & 31;
  const int wave = threadIdx.x >> 5;
  const int half = lane >> 4;
  const int l16  = lane & 15;
  const size_t mbase = (size_t)blockIdx.x * 128 + wave * 16;
  const float* dr = D + (mbase + l16) * DEC0;
  const float4 f0 = *(const float4*)(dr + 8 * half);
  const float4 f1 = *(const float4*)(dr + 8 * half + 4);
  const float4 f2 = *(const float4*)(dr + 16 + 8 * half);
  const float4 f3 = *(const float4*)(dr + 20 + 8 * half);
  v16h a;
  a[0]  = (_Float16)f0.x; a[1]  = (_Float16)f0.y;
  a[2]  = (_Float16)f0.z; a[3]  = (_Float16)f0.w;
  a[4]  = (_Float16)f1.x; a[5]  = (_Float16)f1.y;
  a[6]  = (_Float16)f1.z; a[7]  = (_Float16)f1.w;
  a[8]  = (_Float16)f2.x; a[9]  = (_Float16)f2.y;
  a[10] = (_Float16)f2.z; a[11] = (_Float16)f2.w;
  a[12] = (_Float16)f3.x; a[13] = (_Float16)f3.y;
  a[14] = (_Float16)f3.z; a[15] = (_Float16)f3.w;

  stage_dec_chunk(Wd, &lB[0][0], threadIdx.x);          // prologue: chunk 0
  for (int ci = 0; ci < NCHUNK; ++ci) {
    if (ci + 1 < NCHUNK)                                // overlap: issue chunk ci+1
      stage_dec_chunk(Wd + (size_t)(ci + 1) * CCH * 512, &lB[(ci + 1) & 1][0],
                      threadIdx.x);
#if HAS_TDM
    if (ci + 1 < NCHUNK) __builtin_amdgcn_s_wait_tensorcnt(1);  // chunk ci landed
    else                 __builtin_amdgcn_s_wait_tensorcnt(0);
#else
    WAIT_ASYNC();
#endif
    __syncthreads();
    const _Float16* lb = &lB[ci & 1][0];
    for (int ctl = 0; ctl < CCH; ++ctl) {
      const int ct = ci * CCH + ctl;
      const v16h b = *(const v16h*)(lb + (size_t)ctl * 512 + (size_t)lane * 16);
      v8f acc = {};
      acc = __builtin_amdgcn_wmma_f32_16x16x32_f16(false, a, false, b, (short)0, acc, false, false);
      const int c = ct * 16 + l16;
      const float bias = db1[c];
#pragma unroll
      for (int r = 0; r < 8; ++r)
        out_de[(mbase + r + 8 * half) * DIN + c] = acc[r] + bias;
    }
    __syncthreads();
  }
}

// ---------------------------------------------------------------------------
// DEC soft assignment q (student-t, alpha=0.9)
// ---------------------------------------------------------------------------
__global__ void k_q(const float* __restrict__ z, const float* __restrict__ cluster,
                    float* __restrict__ out_q) {
  __shared__ float cl[NC * ZDIM];
  for (int i = threadIdx.x; i < NC * ZDIM; i += blockDim.x) cl[i] = cluster[i];
  __syncthreads();
  int row = blockIdx.x * blockDim.x + threadIdx.x;
  if (row >= N) return;
  float zr[ZDIM];
#pragma unroll
  for (int k = 0; k < ZDIM; ++k) zr[k] = z[(size_t)row * ZDIM + k];
  float q[NC], sum = 0.0f;
#pragma unroll
  for (int j = 0; j < NC; ++j) {
    float sq = 0.0f;
#pragma unroll
    for (int k = 0; k < ZDIM; ++k) {
      float dlt = zr[k] - cl[j * ZDIM + k];
      sq += dlt * dlt;
    }
    float v = powf(1.0f + sq / 0.9f, -0.95f);   // (alpha+1)/2 = 0.95
    q[j] = v; sum += v;
  }
  float rs = 1.0f / sum;
#pragma unroll
  for (int j = 0; j < NC; ++j) out_q[(size_t)row * NC + j] = q[j] * rs;
}

// ---------------------------------------------------------------------------
// Host-side orchestration
// ---------------------------------------------------------------------------
extern "C" void kernel_launch(void* const* d_in, const int* in_sizes, int n_in,
                              void* d_out, int out_size, void* d_ws, size_t ws_size,
                              hipStream_t stream) {
  const float* x    = (const float*)d_in[0];
  const int*   ei   = (const int*)d_in[1];
  const float* eW0  = (const float*)d_in[2];
  const float* eb0  = (const float*)d_in[3];
  const float* eg0  = (const float*)d_in[4];
  const float* ebe0 = (const float*)d_in[5];
  const float* eW1  = (const float*)d_in[6];
  const float* eb1  = (const float*)d_in[7];
  const float* eg1  = (const float*)d_in[8];
  const float* ebe1 = (const float*)d_in[9];
  const float* cW   = (const float*)d_in[10];
  const float* cb   = (const float*)d_in[11];
  const float* cg   = (const float*)d_in[12];
  const float* cbe  = (const float*)d_in[13];
  const float* mW   = (const float*)d_in[14];
  const float* mb   = (const float*)d_in[15];
  const float* lW   = (const float*)d_in[16];
  const float* lb   = (const float*)d_in[17];
  const float* dW0  = (const float*)d_in[18];
  const float* db0  = (const float*)d_in[19];
  const float* dg0  = (const float*)d_in[20];
  const float* dbe0 = (const float*)d_in[21];
  const float* dW1  = (const float*)d_in[22];
  const float* db1  = (const float*)d_in[23];
  const float* clus = (const float*)d_in[24];

  float* out = (float*)d_out;
  float* out_z  = out;
  float* out_mu = out + (size_t)N * 28;
  float* out_lv = out + (size_t)N * 36;
  float* out_de = out + (size_t)N * 44;
  float* out_q  = out + (size_t)N * 2044;
  float* out_fx = out + (size_t)N * 2059;
  float* out_gz = out + (size_t)N * 2079;

  float* w     = (float*)d_ws;
  float* dinv  = w;                          // [N]       (deg -> dinv in place)
  float* H     = w + (size_t)N;              // [N,32]
  float* xw1   = w + (size_t)N * 33;         // [N,64]
  float* cx    = w + (size_t)N * 97;         // [N,64]    (acc -> relu(bn) in place)
  float* xwm   = w + (size_t)N * 161;        // [N,8]
  float* xwl   = w + (size_t)N * 169;        // [N,8]
  float* dbuf  = w + (size_t)N * 177;        // [N,32]
  _Float16* Wb = (_Float16*)(w + (size_t)N * 209);   // 64512 halves (packed enc W)
  _Float16* Wd = Wb + 65536;                          // 64000 halves (packed dec W)

  const int T = 256;
  // graph normalization
  k_deg_init<<<N / T, T, 0, stream>>>(dinv);
  k_deg_acc <<<E / T, T, 0, stream>>>(ei, dinv);
  k_dinv    <<<N / T, T, 0, stream>>>(dinv);
  // weight packing for WMMA
  k_prep_enc_w<<<(KB_ENC * 2 * 32 * 16 + T - 1) / T, T, 0, stream>>>(eW0, Wb);
  k_prep_dec_w<<<(CT_DEC * 32 * 16 + T - 1) / T, T, 0, stream>>>(dW1, Wd);
  // encoder
  k_gemm_enc<<<N / 128, T, 0, stream>>>(x, Wb, eb0, eg0, ebe0, H);
  k_enc2   <<<N / T, T, 0, stream>>>(H, eW1, eb1, eg1, ebe1, out_fx, out_z);
  // GCN conv1
  k_xw1     <<<N * 4 / T, T, 0, stream>>>(out_fx, cW, xw1);
  k_gcn_init<<<N * CONV0 / T, T, 0, stream>>>(xw1, dinv, cx);
  k_gcn_edge<<<E / T, T, 0, stream>>>(ei, dinv, xw1, cx);
  k_gcn_fin <<<N * CONV0 / T, T, 0, stream>>>(cx, cb, cg, cbe);
  // GCN mu / logvar heads
  k_heads    <<<N / T, T, 0, stream>>>(cx, mW, lW, xwm, xwl);
  k_head_init<<<N * ZG / T, T, 0, stream>>>(xwm, xwl, dinv, out_mu, out_lv);
  k_head_edge<<<E / T, T, 0, stream>>>(ei, dinv, xwm, xwl, out_mu, out_lv);
  k_head_fin <<<N / T, T, 0, stream>>>(out_mu, out_lv, mb, lb, out_gz, out_z);
  // decoder
  k_dec0    <<<N * DEC0 / T, T, 0, stream>>>(out_z, dW0, db0, dg0, dbe0, dbuf);
  k_gemm_dec<<<N / 128, T, 0, stream>>>(dbuf, Wd, db1, out_de);
  // DEC soft assignment
  k_q<<<N / T, T, 0, stream>>>(out_z, clus, out_q);
}
